// MarkdownModel_63127429317072
// MI455X (gfx1250) — compile-verified, matrix operand order checked by reading, not run on hardware
//
#include <hip/hip_runtime.h>
#include <math.h>

typedef __bf16 bf16;
typedef __bf16 v16bf __attribute__((ext_vector_type(16)));
typedef __bf16 v8bf  __attribute__((ext_vector_type(8)));
typedef float  v8f   __attribute__((ext_vector_type(8)));

#define D_MODEL 768
#define FF_DIM  3072
#define NHEAD   12
#define HEADD   64
#define NLAYER  12
#define SEQ     2048
#define BATCH   2

// ---------------------------------------------------------------------------
// WMMA helpers (CDNA5: V_WMMA_F32_16X16X32_BF16, wave32)
// ---------------------------------------------------------------------------
__device__ __forceinline__ v8f wmma_bf16(v16bf a, v16bf b, v8f c) {
  return __builtin_amdgcn_wmma_f32_16x16x32_bf16(
      /*neg_a=*/false, a, /*neg_b=*/false, b,
      /*c_mod=*/(short)0, c, /*reuse_a=*/false, /*reuse_b=*/false);
}

// Load a 16x32 bf16 A-style fragment from a row-major buffer.
// Caller passes: p = base + row*ld + kbase, where row = lane&15,
// kbase = (lane>>4)*8.  Elements i<8 -> K=kbase+i ; i>=8 -> K=kbase+16+(i-8).
__device__ __forceinline__ v16bf load_frag_rm(const bf16* p) {
  v8bf lo = *(const v8bf*)(p);
  v8bf hi = *(const v8bf*)(p + 16);
  v16bf f;
#pragma unroll
  for (int i = 0; i < 8; ++i) { f[i] = lo[i]; f[8 + i] = hi[i]; }
  return f;
}

// ---------------------------------------------------------------------------
// fp32 -> bf16 conversion (weights)
// ---------------------------------------------------------------------------
__global__ void cvt_bf16_kernel(const float* __restrict__ in,
                                bf16* __restrict__ out, size_t n) {
  size_t i = (size_t)blockIdx.x * blockDim.x + threadIdx.x;
  size_t stride = (size_t)gridDim.x * blockDim.x;
  for (; i < n; i += stride) out[i] = (bf16)in[i];
}

// ---------------------------------------------------------------------------
// Tiled WMMA GEMM:  C[M,N] = A[M,K](bf16) * W[K,N](bf16) + bias
// Block = 256 threads (8 waves), block tile 128x128, wave tile 32x64
// (2x4 WMMA frags -> 8 wmma per 32-K step), double-buffered LDS staging
// (1 barrier per K-step, global loads overlap WMMA).
// Output: f32 or bf16 (template); optional exact GELU (template).
// ---------------------------------------------------------------------------
#define TM 128
#define TN 128
#define TK 32
#define LDS_S 40   // row stride (32 + 8 pad), bf16 elements

template <int GELU, int OUT_BF16>
__global__ __launch_bounds__(256) void gemm_bf16_kernel(
    const bf16* __restrict__ A, const bf16* __restrict__ W,
    const float* __restrict__ bias, float* __restrict__ Cf,
    bf16* __restrict__ Cb, int M, int N, int K) {
  __shared__ bf16 As[2][TM * LDS_S];  // [buf][m][k]
  __shared__ bf16 Bs[2][TN * LDS_S];  // [buf][n][k] (transposed)

  const int tid  = threadIdx.x;
  const int lane = tid & 31;
  const int wave = tid >> 5;
  const int wm   = (wave & 3) * 32;   // 0,32,64,96
  const int wn   = (wave >> 2) * 64;  // 0,64
  const int m0   = blockIdx.x * TM;
  const int n0   = blockIdx.y * TN;

  const int r  = lane & 15;
  const int kb = (lane >> 4) * 8;

  // staging assignments: 16 bf16 per thread for each of A and B
  const int arow = tid >> 1;          // 0..127 (m)
  const int acol = (tid & 1) * 16;    // 0,16   (k)
  const int brow = tid >> 3;          // 0..31  (k)
  const int bcol = (tid & 7) * 16;    // 0..112 (n)

  const bf16* aptr = A + (size_t)(m0 + arow) * K + acol;
  const bf16* bptr = W + (size_t)brow * N + n0 + bcol;
  const size_t bstep = (size_t)TK * N;

  // ---- prologue: fetch + stage K-tile 0 into buffer 0
  v8bf ar0 = *(const v8bf*)(aptr);
  v8bf ar1 = *(const v8bf*)(aptr + 8);
  v8bf br0 = *(const v8bf*)(bptr);
  v8bf br1 = *(const v8bf*)(bptr + 8);
  *(v8bf*)(As[0] + arow * LDS_S + acol)     = ar0;
  *(v8bf*)(As[0] + arow * LDS_S + acol + 8) = ar1;
#pragma unroll
  for (int j = 0; j < 8; ++j) {
    Bs[0][(bcol + j) * LDS_S + brow]     = br0[j];
    Bs[0][(bcol + 8 + j) * LDS_S + brow] = br1[j];
  }
  __syncthreads();

  v8f acc[2][4];
#pragma unroll
  for (int i = 0; i < 2; ++i)
#pragma unroll
    for (int j = 0; j < 4; ++j)
#pragma unroll
      for (int e = 0; e < 8; ++e) acc[i][j][e] = 0.f;

  int cur = 0;
  for (int k0 = 0; k0 < K; k0 += TK) {
    const bool more = (k0 + TK) < K;
    // ---- issue global fetch of next K-tile (overlaps with WMMA below)
    if (more) {
      const bf16* an = aptr + (k0 + TK);
      const bf16* bn = bptr + (size_t)(k0 / TK + 1) * bstep;
      ar0 = *(const v8bf*)(an);
      ar1 = *(const v8bf*)(an + 8);
      br0 = *(const v8bf*)(bn);
      br1 = *(const v8bf*)(bn + 8);
      __builtin_prefetch(an + TK, 0, 0);  // global_prefetch_b8: K-tile k0+64
    }
    // ---- compute from current buffer: 8 WMMAs per K-step per wave
    v16bf af[2], bfg[4];
#pragma unroll
    for (int i = 0; i < 2; ++i)
      af[i] = load_frag_rm(As[cur] + (wm + i * 16 + r) * LDS_S + kb);
#pragma unroll
    for (int j = 0; j < 4; ++j)
      bfg[j] = load_frag_rm(Bs[cur] + (wn + j * 16 + r) * LDS_S + kb);
#pragma unroll
    for (int i = 0; i < 2; ++i)
#pragma unroll
      for (int j = 0; j < 4; ++j)
        acc[i][j] = wmma_bf16(af[i], bfg[j], acc[i][j]);
    // ---- stage next tile into alternate buffer
    if (more) {
      int nxt = cur ^ 1;
      *(v8bf*)(As[nxt] + arow * LDS_S + acol)     = ar0;
      *(v8bf*)(As[nxt] + arow * LDS_S + acol + 8) = ar1;
#pragma unroll
      for (int j = 0; j < 8; ++j) {
        Bs[nxt][(bcol + j) * LDS_S + brow]     = br0[j];
        Bs[nxt][(bcol + 8 + j) * LDS_S + brow] = br1[j];
      }
    }
    __syncthreads();
    cur ^= 1;
  }

  // ---- epilogue. C-frag: lane<16 -> N=lane, M=vgpr ; lane>=16 -> M=vgpr+8
  const int g = lane >> 4;
#pragma unroll
  for (int i = 0; i < 2; ++i)
#pragma unroll
    for (int j = 0; j < 4; ++j) {
      int n = n0 + wn + j * 16 + r;
      float bia = bias[n];
#pragma unroll
      for (int rr = 0; rr < 8; ++rr) {
        int m = m0 + wm + i * 16 + rr + g * 8;
        float vv = acc[i][j][rr] + bia;
        if (GELU) vv = 0.5f * vv * (1.0f + erff(vv * 0.70710678118654752f));
        if (OUT_BF16) Cb[(size_t)m * N + n] = (bf16)vv;
        else          Cf[(size_t)m * N + n] = vv;
      }
    }
}

// ---------------------------------------------------------------------------
// Embedding + LayerNorm.  One block per token (256 threads, 3 elems/thread).
// ---------------------------------------------------------------------------
__global__ __launch_bounds__(256) void embed_ln_kernel(
    const int* __restrict__ ids, const float* __restrict__ word_emb,
    const float* __restrict__ pos_emb, const float* __restrict__ type_emb,
    const float* __restrict__ gam, const float* __restrict__ bet,
    float* __restrict__ x, bf16* __restrict__ xb) {
  __shared__ float red[256];
  int token = blockIdx.x;
  int s = token % SEQ;
  int id = ids[token];
  float vals[3];
  float sum = 0.f;
#pragma unroll
  for (int e = 0; e < 3; ++e) {
    int d = threadIdx.x + e * 256;
    float vv = word_emb[(size_t)id * D_MODEL + d] +
               pos_emb[(size_t)(s + 2) * D_MODEL + d] + type_emb[d];
    vals[e] = vv;
    sum += vv;
  }
  red[threadIdx.x] = sum; __syncthreads();
  for (int st = 128; st > 0; st >>= 1) {
    if (threadIdx.x < st) red[threadIdx.x] += red[threadIdx.x + st];
    __syncthreads();
  }
  float mu = red[0] * (1.0f / D_MODEL);
  __syncthreads();
  float vs = 0.f;
#pragma unroll
  for (int e = 0; e < 3; ++e) { float dd = vals[e] - mu; vs += dd * dd; }
  red[threadIdx.x] = vs; __syncthreads();
  for (int st = 128; st > 0; st >>= 1) {
    if (threadIdx.x < st) red[threadIdx.x] += red[threadIdx.x + st];
    __syncthreads();
  }
  float inv = rsqrtf(red[0] * (1.0f / D_MODEL) + 1e-5f);
#pragma unroll
  for (int e = 0; e < 3; ++e) {
    int d = threadIdx.x + e * 256;
    float y = (vals[e] - mu) * inv * gam[d] + bet[d];
    x[(size_t)token * D_MODEL + d]  = y;
    xb[(size_t)token * D_MODEL + d] = (bf16)y;
  }
}

// x = LN(x + t) ; also write bf16 copy
__global__ __launch_bounds__(256) void add_ln_kernel(
    float* __restrict__ x, const float* __restrict__ t,
    const float* __restrict__ gam, const float* __restrict__ bet,
    bf16* __restrict__ xb) {
  __shared__ float red[256];
  int token = blockIdx.x;
  float vals[3];
  float sum = 0.f;
#pragma unroll
  for (int e = 0; e < 3; ++e) {
    int d = threadIdx.x + e * 256;
    size_t idx = (size_t)token * D_MODEL + d;
    float vv = x[idx] + t[idx];
    vals[e] = vv;
    sum += vv;
  }
  red[threadIdx.x] = sum; __syncthreads();
  for (int st = 128; st > 0; st >>= 1) {
    if (threadIdx.x < st) red[threadIdx.x] += red[threadIdx.x + st];
    __syncthreads();
  }
  float mu = red[0] * (1.0f / D_MODEL);
  __syncthreads();
  float vs = 0.f;
#pragma unroll
  for (int e = 0; e < 3; ++e) { float dd = vals[e] - mu; vs += dd * dd; }
  red[threadIdx.x] = vs; __syncthreads();
  for (int st = 128; st > 0; st >>= 1) {
    if (threadIdx.x < st) red[threadIdx.x] += red[threadIdx.x + st];
    __syncthreads();
  }
  float inv = rsqrtf(red[0] * (1.0f / D_MODEL) + 1e-5f);
#pragma unroll
  for (int e = 0; e < 3; ++e) {
    int d = threadIdx.x + e * 256;
    size_t idx = (size_t)token * D_MODEL + d;
    float y = (vals[e] - mu) * inv * gam[d] + bet[d];
    x[idx]  = y;
    xb[idx] = (bf16)y;
  }
}

// ---------------------------------------------------------------------------
// Sliding-window (banded) attention, flash style, WMMA for QK^T and PV.
// q,k,v,out are [B,S,D] bf16 with head h at column h*64.  Window = +-half.
// Block = 128 threads (4 waves); each wave owns one 16-query tile.
// ---------------------------------------------------------------------------
__global__ __launch_bounds__(128) void attn_kernel(
    const bf16* __restrict__ q, const bf16* __restrict__ k,
    const bf16* __restrict__ v, const float* __restrict__ mask,
    bf16* __restrict__ out, int half) {
  __shared__ bf16 pbuf[4][16 * 40];  // per-wave P tile [16][32] (+pad)

  const int lane = threadIdx.x & 31;
  const int wid  = threadIdx.x >> 5;
  const int qt   = blockIdx.x * 4 + wid;
  const int h    = blockIdx.y;
  const int b    = blockIdx.z;
  const int q0   = qt * 16;

  const size_t bofs = (size_t)b * SEQ * D_MODEL + h * HEADD;
  const bf16* qp = q + bofs;
  const bf16* kp = k + bofs;
  const bf16* vp = v + bofs;

  const int r  = lane & 15;
  const int g  = lane >> 4;
  const int kb = g * 8;

  // Q fragments: A = Q[16 x 64], split into hd 0..31 / 32..63
  v16bf aq[2];
#pragma unroll
  for (int kk = 0; kk < 2; ++kk)
    aq[kk] = load_frag_rm(qp + (size_t)(q0 + r) * D_MODEL + kk * 32 + kb);

  v8f acc[4];
  float m[8], l[8];
#pragma unroll
  for (int nf = 0; nf < 4; ++nf)
#pragma unroll
    for (int e = 0; e < 8; ++e) acc[nf][e] = 0.f;
#pragma unroll
  for (int rr = 0; rr < 8; ++rr) { m[rr] = -1e30f; l[rr] = 0.f; }

  int ks0  = (q0 - half) & ~31;   // floor to 32 (negative-safe)
  int kend = q0 + 16 + half;      // exclusive bound of needed keys
  for (int kt = ks0; kt < kend; kt += 32) {
    // ---- K^T B-fragments: [64 hd x 32 keys]; per lane fixed key, hd varies
    v16bf bk[2][2];
#pragma unroll
    for (int nf = 0; nf < 2; ++nf) {
      int key = kt + nf * 16 + r;
      int kc  = min(max(key, 0), SEQ - 1);
#pragma unroll
      for (int kk = 0; kk < 2; ++kk)
        bk[nf][kk] = load_frag_rm(kp + (size_t)kc * D_MODEL + kk * 32 + kb);
    }
    // ---- scores = Q * K^T (two 16x16 frags over 32 keys)
    v8f sc[2];
#pragma unroll
    for (int nf = 0; nf < 2; ++nf) {
#pragma unroll
      for (int e = 0; e < 8; ++e) sc[nf][e] = 0.f;
      sc[nf] = wmma_bf16(aq[0], bk[nf][0], sc[nf]);
      sc[nf] = wmma_bf16(aq[1], bk[nf][1], sc[nf]);
    }
    // ---- band mask + key_add
    float sval[2][8];
    bool  okm[2][8];
#pragma unroll
    for (int nf = 0; nf < 2; ++nf) {
      int col = kt + nf * 16 + r;  // global key index for this lane
      bool colin = (col >= 0) && (col < SEQ);
      int colc = min(max(col, 0), SEQ - 1);
      float ka = (1.0f - mask[(size_t)b * SEQ + colc]) * -1e9f;
#pragma unroll
      for (int rr = 0; rr < 8; ++rr) {
        int row = q0 + rr + g * 8;  // global query index
        bool o = colin && (col >= row - half) && (col <= row + half);
        okm[nf][rr]  = o;
        sval[nf][rr] = o ? (sc[nf][rr] * 0.125f + ka) : -1e30f;
      }
    }
    // ---- online softmax (row reductions across 16-lane half groups)
#pragma unroll
    for (int rr = 0; rr < 8; ++rr) {
      float tmax = fmaxf(sval[0][rr], sval[1][rr]);
#pragma unroll
      for (int sh = 8; sh >= 1; sh >>= 1)
        tmax = fmaxf(tmax, __shfl_xor(tmax, sh, 32));
      float mnew = fmaxf(m[rr], tmax);
      float corr = __expf(m[rr] - mnew);
      float p0 = okm[0][rr] ? __expf(sval[0][rr] - mnew) : 0.f;
      float p1 = okm[1][rr] ? __expf(sval[1][rr] - mnew) : 0.f;
      float ps = p0 + p1;
#pragma unroll
      for (int sh = 8; sh >= 1; sh >>= 1) ps += __shfl_xor(ps, sh, 32);
      l[rr] = l[rr] * corr + ps;
      m[rr] = mnew;
#pragma unroll
      for (int nf = 0; nf < 4; ++nf) acc[nf][rr] *= corr;
      // C-layout -> LDS (row-major P[16][32]) for A-layout reload
      int prow = rr + g * 8;
      pbuf[wid][prow * 40 + r]      = (bf16)p0;
      pbuf[wid][prow * 40 + 16 + r] = (bf16)p1;
    }
    // ---- P as A-fragment [16 q x 32 keys]
    v16bf pf = load_frag_rm(pbuf[wid] + r * 40 + kb);
    // ---- V B-fragments [32 keys x 64 hd] and PV accumulate
#pragma unroll
    for (int nf = 0; nf < 4; ++nf) {
      v16bf bvf;
      int hd = nf * 16 + r;
      int kbase = kt + g * 8;
#pragma unroll
      for (int i = 0; i < 8; ++i) {
        int k1 = min(max(kbase + i, 0), SEQ - 1);
        int k2 = min(max(kbase + 16 + i, 0), SEQ - 1);
        bvf[i]     = vp[(size_t)k1 * D_MODEL + hd];
        bvf[8 + i] = vp[(size_t)k2 * D_MODEL + hd];
      }
      acc[nf] = wmma_bf16(pf, bvf, acc[nf]);
    }
  }

  // ---- normalize and write out[b, s, h*64+hd] (bf16)
#pragma unroll
  for (int nf = 0; nf < 4; ++nf) {
    int hd = nf * 16 + r;
#pragma unroll
    for (int rr = 0; rr < 8; ++rr) {
      int row = q0 + rr + g * 8;
      float invl = (l[rr] > 0.f) ? (1.0f / l[rr]) : 0.f;
      out[(size_t)b * SEQ * D_MODEL + (size_t)row * D_MODEL + h * HEADD + hd] =
          (bf16)(acc[nf][rr] * invl);
    }
  }
}

// ---------------------------------------------------------------------------
// Head: out[b] = concat(cls, fts[b]) @ Wtop + btop
// ---------------------------------------------------------------------------
__global__ __launch_bounds__(256) void top_kernel(
    const float* __restrict__ x, const float* __restrict__ fts,
    const float* __restrict__ Wtop, const float* __restrict__ btop,
    float* __restrict__ out) {
  __shared__ float red[256];
  int b = blockIdx.x;
  float s = 0.f;
  for (int d = threadIdx.x; d < D_MODEL; d += 256)
    s += x[(size_t)b * SEQ * D_MODEL + d] * Wtop[d];
  red[threadIdx.x] = s; __syncthreads();
  for (int st = 128; st > 0; st >>= 1) {
    if (threadIdx.x < st) red[threadIdx.x] += red[threadIdx.x + st];
    __syncthreads();
  }
  if (threadIdx.x == 0) out[b] = red[0] + fts[b] * Wtop[D_MODEL] + btop[0];
}

// ---------------------------------------------------------------------------
// Host launcher
// ---------------------------------------------------------------------------
extern "C" void kernel_launch(void* const* d_in, const int* in_sizes, int n_in,
                              void* d_out, int out_size, void* d_ws,
                              size_t ws_size, hipStream_t stream) {
  (void)in_sizes; (void)n_in; (void)out_size; (void)ws_size;
  const int*   ids      = (const int*)  d_in[0];
  const float* mask     = (const float*)d_in[1];
  const float* fts      = (const float*)d_in[2];
  const float* word_emb = (const float*)d_in[3];
  const float* pos_emb  = (const float*)d_in[4];
  const float* type_emb = (const float*)d_in[5];
  const float* ln_emb_g = (const float*)d_in[6];
  const float* ln_emb_b = (const float*)d_in[7];
  const float* Wq = (const float*)d_in[8];
  const float* bq = (const float*)d_in[9];
  const float* Wk = (const float*)d_in[10];
  const float* bk = (const float*)d_in[11];
  const float* Wv = (const float*)d_in[12];
  const float* bv = (const float*)d_in[13];
  const float* Wo = (const float*)d_in[14];
  const float* bo = (const float*)d_in[15];
  const float* ln1_g = (const float*)d_in[16];
  const float* ln1_b = (const float*)d_in[17];
  const float* W1 = (const float*)d_in[18];
  const float* b1 = (const float*)d_in[19];
  const float* W2 = (const float*)d_in[20];
  const float* b2 = (const float*)d_in[21];
  const float* ln2_g = (const float*)d_in[22];
  const float* ln2_b = (const float*)d_in[23];
  const float* Wtop = (const float*)d_in[24];
  const float* btop = (const float*)d_in[25];

  char* ws = (char*)d_ws;
  size_t off = 0;
  auto alloc = [&](size_t bytes) -> char* {
    char* p = ws + off;
    off += (bytes + 255) & ~(size_t)255;
    return p;
  };
  const size_t DD  = (size_t)D_MODEL * D_MODEL;
  const size_t DF  = (size_t)D_MODEL * FF_DIM;
  const size_t TOK = (size_t)BATCH * SEQ;

  bf16* wqb = (bf16*)alloc(NLAYER * DD * 2);
  bf16* wkb = (bf16*)alloc(NLAYER * DD * 2);
  bf16* wvb = (bf16*)alloc(NLAYER * DD * 2);
  bf16* wob = (bf16*)alloc(NLAYER * DD * 2);
  bf16* w1b = (bf16*)alloc(NLAYER * DF * 2);
  bf16* w2b = (bf16*)alloc(NLAYER * DF * 2);
  float* x    = (float*)alloc(TOK * D_MODEL * 4);
  bf16*  xb   = (bf16*) alloc(TOK * D_MODEL * 2);
  bf16*  qbuf = (bf16*) alloc(TOK * D_MODEL * 2);
  bf16*  kbuf = (bf16*) alloc(TOK * D_MODEL * 2);
  bf16*  vbuf = (bf16*) alloc(TOK * D_MODEL * 2);
  bf16*  abuf = (bf16*) alloc(TOK * D_MODEL * 2);
  bf16*  hbuf = (bf16*) alloc(TOK * FF_DIM * 2);
  float* tbuf = (float*)alloc(TOK * D_MODEL * 4);

  // ---- weights -> bf16 (all bf16 weights ~170MB: fit in the 192MB L2)
  cvt_bf16_kernel<<<2048, 256, 0, stream>>>(Wq, wqb, NLAYER * DD);
  cvt_bf16_kernel<<<2048, 256, 0, stream>>>(Wk, wkb, NLAYER * DD);
  cvt_bf16_kernel<<<2048, 256, 0, stream>>>(Wv, wvb, NLAYER * DD);
  cvt_bf16_kernel<<<2048, 256, 0, stream>>>(Wo, wob, NLAYER * DD);
  cvt_bf16_kernel<<<2048, 256, 0, stream>>>(W1, w1b, NLAYER * DF);
  cvt_bf16_kernel<<<2048, 256, 0, stream>>>(W2, w2b, NLAYER * DF);

  // ---- embeddings + LN
  embed_ln_kernel<<<(int)TOK, 256, 0, stream>>>(ids, word_emb, pos_emb,
                                                type_emb, ln_emb_g, ln_emb_b,
                                                x, xb);

  const int WIN[NLAYER] = {32, 32, 64, 64, 128, 128, 256, 256, 512, 512, 512, 512};
  const int M = (int)TOK;
  dim3 gq(M / TM, D_MODEL / TN);   // GEMMs with N = 768  -> (32, 6)
  dim3 g1(M / TM, FF_DIM / TN);    // GEMM with N = 3072 -> (32, 24)
  dim3 ga(SEQ / 64, NHEAD, BATCH); // attention: 4 q-tiles of 16 per block

  for (int i = 0; i < NLAYER; ++i) {
    gemm_bf16_kernel<0, 1><<<gq, 256, 0, stream>>>(
        xb, wqb + (size_t)i * DD, bq + (size_t)i * D_MODEL, nullptr, qbuf,
        M, D_MODEL, D_MODEL);
    gemm_bf16_kernel<0, 1><<<gq, 256, 0, stream>>>(
        xb, wkb + (size_t)i * DD, bk + (size_t)i * D_MODEL, nullptr, kbuf,
        M, D_MODEL, D_MODEL);
    gemm_bf16_kernel<0, 1><<<gq, 256, 0, stream>>>(
        xb, wvb + (size_t)i * DD, bv + (size_t)i * D_MODEL, nullptr, vbuf,
        M, D_MODEL, D_MODEL);

    attn_kernel<<<ga, 128, 0, stream>>>(qbuf, kbuf, vbuf, mask, abuf,
                                        WIN[i] / 2);

    gemm_bf16_kernel<0, 0><<<gq, 256, 0, stream>>>(
        abuf, wob + (size_t)i * DD, bo + (size_t)i * D_MODEL, tbuf, nullptr,
        M, D_MODEL, D_MODEL);
    add_ln_kernel<<<(int)TOK, 256, 0, stream>>>(
        x, tbuf, ln1_g + (size_t)i * D_MODEL, ln1_b + (size_t)i * D_MODEL, xb);

    gemm_bf16_kernel<1, 1><<<g1, 256, 0, stream>>>(
        xb, w1b + (size_t)i * DF, b1 + (size_t)i * FF_DIM, nullptr, hbuf,
        M, FF_DIM, D_MODEL);
    gemm_bf16_kernel<0, 0><<<gq, 256, 0, stream>>>(
        hbuf, w2b + (size_t)i * DF, b2 + (size_t)i * D_MODEL, tbuf, nullptr,
        M, D_MODEL, FF_DIM);
    add_ln_kernel<<<(int)TOK, 256, 0, stream>>>(
        x, tbuf, ln2_g + (size_t)i * D_MODEL, ln2_b + (size_t)i * D_MODEL, xb);
  }

  top_kernel<<<BATCH, 256, 0, stream>>>(x, fts, Wtop, btop, (float*)d_out);
}
